// DNM_Conv_6176162971842
// MI455X (gfx1250) — matrix operand order repaired and checked by reading, not compiled, not run on hardware
//
#include <hip/hip_runtime.h>
#include <hip/hip_fp16.h>

typedef __attribute__((ext_vector_type(16))) _Float16 v16h;
typedef __attribute__((ext_vector_type(8)))  float    v8f;

namespace {
constexpr int   kC = 128, kO = 32, kM = 8, kHW = 256;
constexpr int   kLdsStride = 132;     // 132 % 64 == 4 -> conflict-spread rows
constexpr float kEps = 1e-5f;
}

__device__ __forceinline__ void unpack4(float* d, float4 v) {
  d[0] = v.x; d[1] = v.y; d[2] = v.z; d[3] = v.w;
}

__global__ __launch_bounds__(32)
void dnm_conv_kernel(const float* __restrict__ x,
                     const float* __restrict__ W,
                     const float* __restrict__ q,
                     const float* __restrict__ qsp,
                     const float* __restrict__ g1,
                     const float* __restrict__ b1,
                     const float* __restrict__ g2,
                     const float* __restrict__ b2,
                     float* __restrict__ out) {
  __shared__ float shXN[16 * kLdsStride];  // LN1(x) tile: 16 pos x 128 c
  __shared__ float shD[32 * 8];            // WMMA D transpose buffer
  __shared__ float shAcc[16 * 32];         // [pos][o] accumulator
  const int lane = threadIdx.x;            // wave32
  const int tile = blockIdx.x;             // 128 tiles of 16 positions
  const int b    = tile >> 4;
  const int hw0  = (tile & 15) << 4;
  const int pos  = lane & 15;              // WMMA A-tile row (M)
  const int hi   = lane >> 4;              // which K-half this lane feeds

  // ---- wave-uniform sums of gamma2 / beta2 ----
  float G  = g2[lane] + g2[lane + 32] + g2[lane + 64] + g2[lane + 96];
  float Bs = b2[lane] + b2[lane + 32] + b2[lane + 64] + b2[lane + 96];
  #pragma unroll
  for (int off = 16; off > 0; off >>= 1) {
    G  += __shfl_xor(G, off, 32);
    Bs += __shfl_xor(Bs, off, 32);
  }

  // ---- Phase 1: load x, LayerNorm over C, stage xn into LDS ----
  const float* xb = x + (size_t)b * kC * kHW + hw0 + pos;
  float s1 = 0.f, s2 = 0.f;
  #pragma unroll 4
  for (int j = 0; j < 64; ++j) {
    const int c = hi * 64 + j;
    const float v = xb[(size_t)c * kHW];   // lanes 0..15 contiguous in hw
    s1 += v;
    s2 = fmaf(v, v, s2);
    shXN[pos * kLdsStride + c] = v;
  }
  s1 += __shfl_xor(s1, 16, 32);
  s2 += __shfl_xor(s2, 16, 32);
  const float mu  = s1 * (1.f / kC);
  const float var = fmaf(-mu, mu, s2 * (1.f / kC));
  const float inv = rsqrtf(var + kEps);
  #pragma unroll 4
  for (int j = 0; j < 64; ++j) {
    const int c   = hi * 64 + j;
    const int idx = pos * kLdsStride + c;
    shXN[idx] = fmaf((shXN[idx] - mu) * inv, g1[c], b1[c]);
  }
  // zero the [pos][o] accumulator (bank-conflict-free columns)
  #pragma unroll
  for (int i = 0; i < 16; ++i) shAcc[i * 32 + lane] = 0.f;
  __syncthreads();

  // ---- hoist this lane's xn values for ALL (o,m) iterations into VGPRs ----
  const float4* sh4 = (const float4*)shXN;  // row stride = 33 float4
  float xnv[64];
  #pragma unroll
  for (int ch = 0; ch < 4; ++ch) {
    const int cA = ch * 32 + hi * 8;       // feeds A elements 0..7
    const int cB = cA + 16;                // feeds A elements 8..15
    unpack4(xnv + ch * 16 + 0,  sh4[pos * (kLdsStride / 4) + (cA >> 2)]);
    unpack4(xnv + ch * 16 + 4,  sh4[pos * (kLdsStride / 4) + (cA >> 2) + 1]);
    unpack4(xnv + ch * 16 + 8,  sh4[pos * (kLdsStride / 4) + (cB >> 2)]);
    unpack4(xnv + ch * 16 + 12, sh4[pos * (kLdsStride / 4) + (cB >> 2) + 1]);
  }

  // ---- B matrices (K=32 rows, lane%16 = N, element = K in table order): col0 = 1 -> S1, col1 = gamma2 -> Sg ----
  v16h Bv[4];
  #pragma unroll
  for (int ch = 0; ch < 4; ++ch) {
    v16h t;
    #pragma unroll
    for (int e = 0; e < 16; ++e) {
      const int k = (e < 8) ? (hi * 8 + e) : (16 + hi * 8 + (e - 8));
      t[e] = (pos == 0) ? (_Float16)1.0f
           : (pos == 1) ? (_Float16)g2[ch * 32 + k] : (_Float16)0.0f;
    }
    Bv[ch] = t;
  }

  const float qsv = qsp[0];
  const float4* W4 = (const float4*)W;
  const float4* q4 = (const float4*)q;
  float4* shD4 = (float4*)shD;
  const int row  = lane & 15;              // stats row this lane evaluates
  const int sl   = (row < 8) ? 0 : 16;     // source lane of S1 in D layout
  const int iS1  = (sl + 0) * 8 + (row & 7);
  const int iSg  = (sl + 1) * 8 + (row & 7);

  for (int o = 0; o < kO; ++o) {
    if (o + 1 < kO) {                      // uniform branch, EXEC stays full
      __builtin_prefetch(W + (size_t)(o + 1) * kM * kC + lane * 32, 0, 1);
      __builtin_prefetch(q + (size_t)(o + 1) * kM * kC + lane * 32, 0, 1);
    }
    float accO = 0.f;                      // sum over m of this row's zn-sum
    #pragma unroll 1
    for (int m = 0; m < kM; ++m) {
      v8f   D    = {};
      float zs2a = 0.f, zs2b = 0.f;        // twin chains -> dual-issued FMAs
      const int wq4 = ((o * kM + m) * kC) >> 2;  // base in float4 units
      #pragma unroll
      for (int ch = 0; ch < 4; ++ch) {
        const int cA = ch * 32 + hi * 8;
        const int cB = cA + 16;
        float wv[16], qv[16];
        unpack4(wv + 0,  W4[wq4 + (cA >> 2)]);
        unpack4(wv + 4,  W4[wq4 + (cA >> 2) + 1]);
        unpack4(wv + 8,  W4[wq4 + (cB >> 2)]);
        unpack4(wv + 12, W4[wq4 + (cB >> 2) + 1]);
        unpack4(qv + 0,  q4[wq4 + (cA >> 2)]);
        unpack4(qv + 4,  q4[wq4 + (cA >> 2) + 1]);
        unpack4(qv + 8,  q4[wq4 + (cB >> 2)]);
        unpack4(qv + 12, q4[wq4 + (cB >> 2) + 1]);
        v16h A;
        #pragma unroll
        for (int e = 0; e < 16; ++e) {
          float z = fmaf(xnv[ch * 16 + e], wv[e], -qv[e]);  // xn*W - q
          z = fmaxf(z, 0.f);                                // relu
          if (e & 1) zs2b = fmaf(z, z, zs2b);
          else       zs2a = fmaf(z, z, zs2a);
          A[e] = (_Float16)z;
        }
        // D[i][0] += sum_k z[i][k];  D[i][1] += sum_k gamma2[k]*z[i][k]
        D = __builtin_amdgcn_wmma_f32_16x16x32_f16(
                false, A, false, Bv[ch], (short)0, D, false, false);
      }
      float zs2 = zs2a + zs2b;
      zs2 += __shfl_xor(zs2, 16, 32);      // lane r (and r+16) hold row r's S2

      // transpose D through LDS: each lane dumps its 8 D floats, then lane i
      // (and i+16, redundantly -> EXEC stays full) picks up row i's S1/Sg.
      shD4[lane * 2 + 0] = make_float4(D[0], D[1], D[2], D[3]);
      shD4[lane * 2 + 1] = make_float4(D[4], D[5], D[6], D[7]);
      __builtin_amdgcn_wave_barrier();     // codegen ordering; DS is in-order
      const float S1 = shD[iS1];
      const float Sg = shD[iSg];
      const float mz = S1 * (1.f / kC);
      const float vz = fmaf(-mz, mz, zs2 * (1.f / kC));
      const float iz = rsqrtf(vz + kEps);
      accO += fmaf(fmaf(-mz, G, Sg), iz, Bs);   // sum_c zn for (row, o, m)
    }
    // lanes i and i+16 write identical values to the same address: benign.
    shAcc[row * 32 + o] = accO;
  }
  __syncthreads();

  // ---- final transpose read: lane == o, 16 consecutive hw outputs ----
  float res[16];
  #pragma unroll
  for (int i = 0; i < 16; ++i) res[i] = shAcc[i * 32 + lane];  // conflict-free
  float4* out4 = (float4*)out;
  const int obase = (((b * kO + lane) * kHW) + hw0) >> 2;
  #pragma unroll
  for (int j2 = 0; j2 < 4; ++j2) {
    float4 r;
    r.x = fmaxf(res[4 * j2 + 0] - qsv, 0.f);
    r.y = fmaxf(res[4 * j2 + 1] - qsv, 0.f);
    r.z = fmaxf(res[4 * j2 + 2] - qsv, 0.f);
    r.w = fmaxf(res[4 * j2 + 3] - qsv, 0.f);
    out4[obase + j2] = r;
  }
}

extern "C" void kernel_launch(void* const* d_in, const int* in_sizes, int n_in,
                              void* d_out, int out_size, void* d_ws, size_t ws_size,
                              hipStream_t stream) {
  (void)in_sizes; (void)n_in; (void)out_size; (void)d_ws; (void)ws_size;
  const float* x  = (const float*)d_in[0];
  const float* W  = (const float*)d_in[1];
  const float* q  = (const float*)d_in[2];
  const float* qs = (const float*)d_in[3];
  const float* g1 = (const float*)d_in[4];
  const float* b1 = (const float*)d_in[5];
  const float* g2 = (const float*)d_in[6];
  const float* b2 = (const float*)d_in[7];
  float* out = (float*)d_out;
  // 128 tiles (2048 positions / 16), one wave32 per tile
  dnm_conv_kernel<<<128, 32, 0, stream>>>(x, W, q, qs, g1, b1, g2, b2, out);
}
